// LoraMoeDecoderLayer_9474697855507
// MI455X (gfx1250) — matrix operand order repaired and checked by reading, not compiled, not run on hardware
//
#include <hip/hip_runtime.h>
#include <hip/hip_bf16.h>
#include <cstdint>

// ---------------------------------------------------------------------------
// LoraMoeDecoderLayer for MI455X (gfx1250):
//   * wave32 WMMA bf16 (v_wmma_f32_16x16x32_bf16) for all GEMMs + attention
//   * TDM (tensor_load_to_lds / TENSORcnt) staging for GEMM tiles
//   * async Global->LDS (ASYNCcnt) staging in flash attention
// ---------------------------------------------------------------------------

typedef __bf16 bf16;
typedef uint32_t u32;
typedef uint64_t u64;
typedef __attribute__((ext_vector_type(16))) __bf16 v16bf;
typedef __attribute__((ext_vector_type(8)))  __bf16 v8bf;
typedef __attribute__((ext_vector_type(8)))  float  v8f;
typedef __attribute__((ext_vector_type(4)))  u32    su32x4;
typedef __attribute__((ext_vector_type(8)))  u32    su32x8;

namespace cfg {
constexpr int B = 2, S = 2048, D = 2048, H = 16, KV = 8, HD = 128;
constexpr int E = 8, R = 16, FF = 5632;
constexpr int T = B * S;
constexpr float LORA_SCALE = 2.0f;   // alpha/rank = 32/16
constexpr float EPS = 1e-6f;
constexpr float SCALE = 0.08838834764831845f;  // HD^-0.5
}

// ------------------------------- WMMA helpers ------------------------------

__device__ __forceinline__ v8f wmma_bf16(v16bf a, v16bf b, v8f c) {
  // D = A(16x32) * B(32x16) + C, f32 accum
  return __builtin_amdgcn_wmma_f32_16x16x32_bf16(false, a, false, b,
                                                 (short)0, c, false, false);
}

// A-fragment: lane holds row (lane%16); k-halves: [khalf, khalf+8) and
// [16+khalf, 16+khalf+8) where khalf = (lane/16)*8.  Source row-major over K.
__device__ __forceinline__ v16bf load_frag_a(const bf16* row, int lhi) {
  union { v16bf v; v8bf h[2]; } u;
  u.h[0] = *(const v8bf*)(row + lhi * 8);
  u.h[1] = *(const v8bf*)(row + 16 + lhi * 8);
  return u.v;
}

// B-fragment: lane holds column (lane%16); k-halves: [khalf, khalf+8) and
// [16+khalf, 16+khalf+8) where khalf = (lane/16)*8.
__device__ __forceinline__ v16bf load_frag_b(const bf16* row, int lhi) {
  union { v16bf v; v8bf h[2]; } u;
  u.h[0] = *(const v8bf*)(row + lhi * 8);
  u.h[1] = *(const v8bf*)(row + 16 + lhi * 8);
  return u.v;
}

// --------------------- CDNA5 async Global->LDS copy ------------------------
// GLOBAL_LOAD_ASYNC_TO_LDS_B128: per-lane 16B memory -> LDS, no VGPR data,
// tracked by ASYNCcnt.  LDS dest address = low 32 bits of the generic pointer.

__device__ __forceinline__ void async_copy_b128(const void* gaddr, void* lds) {
  unsigned l = (unsigned)(size_t)lds;
  asm volatile("global_load_async_to_lds_b128 %0, %1, off"
               :: "v"(l), "v"(gaddr)
               : "memory");
}

__device__ __forceinline__ void wait_async0() {
  asm volatile("s_wait_asynccnt 0x0" ::: "memory");
}

// ------------------- CDNA5 Tensor Data Mover (TDM) -------------------------
// 2D tile DMA Global->LDS.  D# groups per CDNA5 ISA:
//   group0: [1:0]=count=1 | [63:32]=lds_addr | [120:64]=global_addr | type=2
//   group1: mask=0, data_size=1(2B), tensor_dim0/1, tile_dim0/1, dim0 stride
// Rows are written compacted to LDS => LDS tile layout [tile_d1][tile_d0].

__device__ __forceinline__ void tdm_load_tile_2d(const void* gaddr, void* lds,
                                                 u32 tensor_d0, u32 tensor_d1,
                                                 u32 tile_d0, u32 tile_d1,
                                                 u64 stride0_elems) {
  const u64 ga = (u64)(size_t)gaddr;          // byte address (57-bit)
  const u32 la = (u32)(size_t)lds;            // LDS byte address
  su32x4 g0;
  g0[0] = 1u;                                  // count=1, user-mode load
  g0[1] = la;                                  // lds_addr
  g0[2] = (u32)ga;                             // global_addr[31:0]
  g0[3] = (u32)((ga >> 32) & 0x01FFFFFFu) | (2u << 30);  // addr[56:32]|type=2
  su32x8 g1;
  g1[0] = (1u << 16);                          // workgroup_mask=0, data_size=2B
  g1[1] = (tensor_d0 & 0xFFFFu) << 16;         // atomic_barrier_addr=0
  g1[2] = (tensor_d0 >> 16) | ((tensor_d1 & 0xFFFFu) << 16);
  g1[3] = (tensor_d1 >> 16) | (tile_d0 << 16);
  g1[4] = tile_d1;                             // tile_dim2 = 0 (2D)
  g1[5] = (u32)(stride0_elems & 0xFFFFFFFFu);  // tensor_dim0_stride[31:0]
  g1[6] = (u32)((stride0_elems >> 32) & 0xFFFFu);  // stride0[47:32], stride1=0
  g1[7] = 0u;
  asm volatile("tensor_load_to_lds %0, %1" :: "s"(g0), "s"(g1) : "memory");
}

__device__ __forceinline__ void wait_tensor0() {
  __builtin_amdgcn_s_wait_tensorcnt((short)0);
}

// ------------------------- Generic bf16 WMMA GEMM --------------------------
// C[M,N] = A[M,K] @ W[N,K]^T (+bias[n]) (+addend[m,n]);  M%128==N%128==K%32==0
// Tiles staged by the Tensor Data Mover (wave 0 issues, TENSORcnt waits).

__global__ __launch_bounds__(256) void gemm_bf16_wmma(
    const bf16* __restrict__ A, const bf16* __restrict__ W,
    const float* __restrict__ bias, const float* __restrict__ addend,
    float* __restrict__ C, int M, int N, int K) {
  __shared__ __align__(64) bf16 Abuf[2][128 * 32];
  __shared__ __align__(64) bf16 Wbuf[2][128 * 32];
  const int tid  = threadIdx.x;
  const int wave = tid >> 5, lane = tid & 31;
  const int lrow = lane & 15, lhi = lane >> 4;
  const int tM = blockIdx.y * 128, tN = blockIdx.x * 128;
  const int m0 = (wave >> 1) * 32, n0 = (wave & 1) * 64;
  v8f acc[2][4] = {};

  const bf16* Atile = A + (size_t)tM * K;   // tile row base
  const bf16* Wtile = W + (size_t)tN * K;
  const int KT = K >> 5;

  // prologue: TDM-stage tile 0 into buffer 0 (one issuing wave per WG)
  if (wave == 0) {
    tdm_load_tile_2d(Atile, &Abuf[0][0], (u32)K, 128u, 32u, 128u, (u64)K);
    tdm_load_tile_2d(Wtile, &Wbuf[0][0], (u32)K, 128u, 32u, 128u, (u64)K);
  }
  int buf = 0;

  for (int kt = 0; kt < KT; ++kt) {
    if (wave == 0) wait_tensor0();  // DMA for buf complete (wave 0 issued it)
    __syncthreads();                // publish LDS tile to all waves
    const bf16* Ab = Abuf[buf];
    const bf16* Wb = Wbuf[buf];
    if (kt + 1 < KT && wave == 0) {  // TDM-stage next tile into other buffer
      tdm_load_tile_2d(Atile + (size_t)(kt + 1) * 32, &Abuf[buf ^ 1][0],
                       (u32)K, 128u, 32u, 128u, (u64)K);
      tdm_load_tile_2d(Wtile + (size_t)(kt + 1) * 32, &Wbuf[buf ^ 1][0],
                       (u32)K, 128u, 32u, 128u, (u64)K);
    }
    // load all fragments first so ds_loads overlap the WMMA chain
    v16bf af0 = load_frag_a(Ab + (m0 + lrow) * 32, lhi);
    v16bf af1 = load_frag_a(Ab + (m0 + 16 + lrow) * 32, lhi);
    v16bf wf[4];
#pragma unroll
    for (int j = 0; j < 4; ++j)
      wf[j] = load_frag_b(Wb + (n0 + j * 16 + lrow) * 32, lhi);
#pragma unroll
    for (int j = 0; j < 4; ++j) acc[0][j] = wmma_bf16(af0, wf[j], acc[0][j]);
#pragma unroll
    for (int j = 0; j < 4; ++j) acc[1][j] = wmma_bf16(af1, wf[j], acc[1][j]);
    buf ^= 1;
  }

#pragma unroll
  for (int i = 0; i < 2; ++i) {
#pragma unroll
    for (int j = 0; j < 4; ++j) {
      const int ncol = tN + n0 + j * 16 + lrow;
      const float bv = bias ? bias[ncol] : 0.0f;
#pragma unroll
      for (int r = 0; r < 8; ++r) {
        const int row = tM + m0 + i * 16 + lhi * 8 + r;
        float v = acc[i][j][r] + bv;
        if (addend) v += addend[(size_t)row * N + ncol];
        C[(size_t)row * N + ncol] = v;
      }
    }
  }
}

// -------------------------- Flash attention (causal) -----------------------
// grid: (S/128, H, B); block: 256 (8 waves, 16 query rows each)

__global__ __launch_bounds__(256) void flash_attn_wmma(
    const bf16* __restrict__ qh, const bf16* __restrict__ kh,
    const bf16* __restrict__ vh, bf16* __restrict__ attnb) {
  using namespace cfg;
  __shared__ __align__(64) bf16 Klds[32 * 128];   // [key][hd]
  __shared__ __align__(64) bf16 Vt[128 * 32];     // [hd][key]
  __shared__ __align__(64) bf16 pst[8][16 * 32];  // per-wave P staging
  const int tq = blockIdx.x, h = blockIdx.y, b = blockIdx.z;
  const int kvh = h / (H / KV);
  const bf16* Q  = qh + ((size_t)(b * H + h) * S) * HD;
  const bf16* Kp = kh + ((size_t)(b * KV + kvh) * S) * HD;
  const bf16* Vp = vh + ((size_t)(b * KV + kvh) * S) * HD;
  const int tid  = threadIdx.x;
  const int wave = tid >> 5, lane = tid & 31;
  const int lrow = lane & 15, lhi = lane >> 4;
  const int qbase = tq * 128 + wave * 16;

  v16bf qf[4];
#pragma unroll
  for (int s = 0; s < 4; ++s)
    qf[s] = load_frag_a(Q + (size_t)(qbase + lrow) * HD + s * 32, lhi);

  v8f o[8] = {};
  float mrow[8], lsum[8];
#pragma unroll
  for (int r = 0; r < 8; ++r) { mrow[r] = -1e30f; lsum[r] = 0.0f; }

  const int ckey = tid >> 3;        // 0..31
  const int ccol = (tid & 7) * 16;  // 0..112
  const int kend = tq * 128 + 128;

  for (int kc = 0; kc < kend; kc += 32) {
    __syncthreads();
    {  // K chunk: async Global->LDS; V chunk: manual transposed store
      const bf16* src = Kp + (size_t)(kc + ckey) * HD + ccol;
      async_copy_b128(src,     &Klds[ckey * 128 + ccol]);
      async_copy_b128(src + 8, &Klds[ckey * 128 + ccol + 8]);
      const bf16* vs = Vp + (size_t)(kc + ckey) * HD + ccol;
      v8bf d0 = *(const v8bf*)vs;
      v8bf d1 = *(const v8bf*)(vs + 8);
#pragma unroll
      for (int i = 0; i < 8; ++i) {
        Vt[(ccol + i) * 32 + ckey]     = d0[i];
        Vt[(ccol + 8 + i) * 32 + ckey] = d1[i];
      }
    }
    wait_async0();
    __syncthreads();

    // load all K fragments up front, then run the WMMA chain
    v16bf kf[8];
#pragma unroll
    for (int s = 0; s < 4; ++s) {
      kf[s]     = load_frag_b(Klds + lrow * 128 + s * 32, lhi);
      kf[4 + s] = load_frag_b(Klds + (16 + lrow) * 128 + s * 32, lhi);
    }
    v8f sc0 = {}, sc1 = {};
#pragma unroll
    for (int s = 0; s < 4; ++s) {
      sc0 = wmma_bf16(qf[s], kf[s], sc0);
      sc1 = wmma_bf16(qf[s], kf[4 + s], sc1);
    }

    // scale + causal mask + online softmax (per-row stats via 16-lane shfl)
#pragma unroll
    for (int r = 0; r < 8; ++r) {
      const int row = qbase + lhi * 8 + r;
      const int c0 = kc + lrow, c1 = kc + 16 + lrow;
      float s0 = sc0[r] * SCALE;
      float s1 = sc1[r] * SCALE;
      if (c0 > row) s0 = -3.0e38f;
      if (c1 > row) s1 = -3.0e38f;
      float rowm = fmaxf(s0, s1);
      rowm = fmaxf(rowm, __shfl_xor(rowm, 1, 32));
      rowm = fmaxf(rowm, __shfl_xor(rowm, 2, 32));
      rowm = fmaxf(rowm, __shfl_xor(rowm, 4, 32));
      rowm = fmaxf(rowm, __shfl_xor(rowm, 8, 32));
      const float mn  = fmaxf(mrow[r], rowm);
      const float csc = __expf(mrow[r] - mn);
      const float p0  = __expf(s0 - mn);
      const float p1  = __expf(s1 - mn);
      float ps = p0 + p1;
      ps += __shfl_xor(ps, 1, 32);
      ps += __shfl_xor(ps, 2, 32);
      ps += __shfl_xor(ps, 4, 32);
      ps += __shfl_xor(ps, 8, 32);
      lsum[r] = lsum[r] * csc + ps;
      mrow[r] = mn;
#pragma unroll
      for (int f2 = 0; f2 < 8; ++f2) o[f2][r] *= csc;
      pst[wave][(lhi * 8 + r) * 32 + lrow]      = (bf16)p0;
      pst[wave][(lhi * 8 + r) * 32 + 16 + lrow] = (bf16)p1;
    }

    // P (16x32) @ V (32x128): wave-private LDS round trip into A-frag layout
    v16bf pf = load_frag_a(&pst[wave][lrow * 32], lhi);
#pragma unroll
    for (int f2 = 0; f2 < 8; ++f2) {
      v16bf vf = load_frag_b(Vt + (f2 * 16 + lrow) * 32, lhi);
      o[f2] = wmma_bf16(pf, vf, o[f2]);
    }
  }

#pragma unroll
  for (int f2 = 0; f2 < 8; ++f2)
#pragma unroll
    for (int r = 0; r < 8; ++r) {
      const int row = qbase + lhi * 8 + r;
      const float val = o[f2][r] / lsum[r];
      attnb[((size_t)b * S + row) * D + h * HD + f2 * 16 + lrow] = (bf16)val;
    }
}

// ------------------------------ RMSNorm ------------------------------------

__global__ __launch_bounds__(256) void rmsnorm_bf16(
    const float* __restrict__ x, const float* __restrict__ w,
    bf16* __restrict__ out) {
  using namespace cfg;
  const int t = blockIdx.x;
  const float* xr = x + (size_t)t * D;
  float v[8];
  float ss = 0.f;
#pragma unroll
  for (int i = 0; i < 8; ++i) {
    v[i] = xr[threadIdx.x + i * 256];
    ss += v[i] * v[i];
  }
  ss += __shfl_xor(ss, 1, 32);
  ss += __shfl_xor(ss, 2, 32);
  ss += __shfl_xor(ss, 4, 32);
  ss += __shfl_xor(ss, 8, 32);
  ss += __shfl_xor(ss, 16, 32);
  __shared__ float red[9];
  if ((threadIdx.x & 31) == 0) red[threadIdx.x >> 5] = ss;
  __syncthreads();
  if (threadIdx.x == 0) {
    float s = 0.f;
    for (int i = 0; i < 8; ++i) s += red[i];
    red[8] = rsqrtf(s / (float)D + EPS);
  }
  __syncthreads();
  const float r = red[8];
#pragma unroll
  for (int i = 0; i < 8; ++i) {
    const int d = threadIdx.x + i * 256;
    out[(size_t)t * D + d] = (bf16)(v[i] * r * w[d]);
  }
}

// ------------------------- RoPE + layout transforms ------------------------

__global__ void rope_q_kernel(const float* __restrict__ qf, bf16* __restrict__ qh) {
  using namespace cfg;
  const size_t n = (size_t)T * H * (HD / 2);
  size_t i = (size_t)blockIdx.x * blockDim.x + threadIdx.x;
  if (i >= n) return;
  const int half = (int)(i % (HD / 2));
  const int h    = (int)((i / (HD / 2)) % H);
  const int t    = (int)(i / ((size_t)(HD / 2) * H));
  const int s = t % S, b = t / S;
  const float freq = __expf(-((float)(2 * half) / (float)HD) * 13.815510557964274f);
  const float ang = (float)s * freq;
  const float c = cosf(ang), sn = sinf(ang);
  const float x1 = qf[(size_t)t * D + h * HD + half];
  const float x2 = qf[(size_t)t * D + h * HD + half + HD / 2];
  const size_t o = ((size_t)(b * H + h) * S + s) * HD;
  qh[o + half]          = (bf16)(x1 * c - x2 * sn);
  qh[o + half + HD / 2] = (bf16)(x2 * c + x1 * sn);
}

__global__ void rope_k_kernel(const float* __restrict__ kf, bf16* __restrict__ kh) {
  using namespace cfg;
  const size_t n = (size_t)T * KV * (HD / 2);
  size_t i = (size_t)blockIdx.x * blockDim.x + threadIdx.x;
  if (i >= n) return;
  const int half = (int)(i % (HD / 2));
  const int kv   = (int)((i / (HD / 2)) % KV);
  const int t    = (int)(i / ((size_t)(HD / 2) * KV));
  const int s = t % S, b = t / S;
  const float freq = __expf(-((float)(2 * half) / (float)HD) * 13.815510557964274f);
  const float ang = (float)s * freq;
  const float c = cosf(ang), sn = sinf(ang);
  const float x1 = kf[(size_t)t * (KV * HD) + kv * HD + half];
  const float x2 = kf[(size_t)t * (KV * HD) + kv * HD + half + HD / 2];
  const size_t o = ((size_t)(b * KV + kv) * S + s) * HD;
  kh[o + half]          = (bf16)(x1 * c - x2 * sn);
  kh[o + half + HD / 2] = (bf16)(x2 * c + x1 * sn);
}

__global__ void cast_v_kernel(const float* __restrict__ vf, bf16* __restrict__ vh) {
  using namespace cfg;
  const size_t n = (size_t)T * KV * HD;
  size_t i = (size_t)blockIdx.x * blockDim.x + threadIdx.x;
  if (i >= n) return;
  const int hd = (int)(i % HD);
  const int kv = (int)((i / HD) % KV);
  const int t  = (int)(i / (size_t)(KV * HD));
  const int s = t % S, b = t / S;
  vh[((size_t)(b * KV + kv) * S + s) * HD + hd] = (bf16)vf[i];
}

// ------------------------- Router (noisy top-2) ----------------------------

__global__ __launch_bounds__(256) void router_kernel(
    const bf16* __restrict__ x, const float* __restrict__ rw,
    const float* __restrict__ nw, const float* __restrict__ noise,
    float* __restrict__ Wd, float* __restrict__ swsum) {
  using namespace cfg;
  const int t = blockIdx.x;
  const bf16* xr = x + (size_t)t * D;
  float acc[2 * E] = {};
  for (int d = threadIdx.x; d < D; d += 256) {
    const float xv = (float)xr[d];
#pragma unroll
    for (int e = 0; e < E; ++e) {
      acc[e]     += xv * rw[(size_t)e * D + d];
      acc[E + e] += xv * nw[(size_t)e * D + d];
    }
  }
#pragma unroll
  for (int e = 0; e < 2 * E; ++e) {
    acc[e] += __shfl_xor(acc[e], 1, 32);
    acc[e] += __shfl_xor(acc[e], 2, 32);
    acc[e] += __shfl_xor(acc[e], 4, 32);
    acc[e] += __shfl_xor(acc[e], 8, 32);
    acc[e] += __shfl_xor(acc[e], 16, 32);
  }
  __shared__ float red[8][2 * E];
  if ((threadIdx.x & 31) == 0)
#pragma unroll
    for (int e = 0; e < 2 * E; ++e) red[threadIdx.x >> 5][e] = acc[e];
  __syncthreads();
  if (threadIdx.x == 0) {
    float L[2 * E];
    for (int e = 0; e < 2 * E; ++e) {
      float s = 0.f;
      for (int w = 0; w < 8; ++w) s += red[w][e];
      L[e] = s;
    }
    float nl[E];
    for (int e = 0; e < E; ++e) {
      const float xn = L[E + e];
      const float sp = (xn > 20.0f) ? xn : log1pf(__expf(xn));
      nl[e] = L[e] + noise[(size_t)t * E + e] * sp;
    }
    float mx = nl[0];
    for (int e = 1; e < E; ++e) mx = fmaxf(mx, nl[e]);
    float p[E], s = 0.f;
    for (int e = 0; e < E; ++e) { p[e] = __expf(nl[e] - mx); s += p[e]; }
    for (int e = 0; e < E; ++e) p[e] /= s;
    int i1 = 0;
    for (int e = 1; e < E; ++e) if (p[e] > p[i1]) i1 = e;
    int i2 = (i1 == 0) ? 1 : 0;
    for (int e = 0; e < E; ++e) if (e != i1 && p[e] > p[i2]) i2 = e;
    const float tot = p[i1] + p[i2];
    float outw[E] = {};
    outw[i1] = p[i1] / tot;
    outw[i2] = p[i2] / tot;
    for (int e = 0; e < E; ++e) Wd[(size_t)t * E + e] = outw[e];
    swsum[t] = outw[i1] + outw[i2];
  }
}

// ------------------------------ elementwise --------------------------------

__global__ void cvt_f32_to_bf16(const float* __restrict__ in,
                                bf16* __restrict__ out, size_t n) {
  size_t i = (size_t)blockIdx.x * blockDim.x + threadIdx.x;
  const size_t stride = (size_t)gridDim.x * blockDim.x;
  for (; i < n; i += stride) out[i] = (bf16)in[i];
}

__global__ void transpose_lora_b(const float* __restrict__ lb,
                                 bf16* __restrict__ outw) {
  using namespace cfg;
  const size_t n = (size_t)E * D * R;
  size_t i = (size_t)blockIdx.x * blockDim.x + threadIdx.x;
  if (i >= n) return;
  const int r = (int)(i % R);
  const int d = (int)((i / R) % D);
  const int e = (int)(i / ((size_t)R * D));
  outw[(size_t)d * (E * R) + e * R + r] = (bf16)lb[i];
}

__global__ void swiglu_kernel(const float* __restrict__ g,
                              const float* __restrict__ u,
                              bf16* __restrict__ act, size_t n) {
  size_t i = (size_t)blockIdx.x * blockDim.x + threadIdx.x;
  if (i >= n) return;
  const float gv = g[i];
  const float s  = gv / (1.0f + __expf(-gv));
  act[i] = (bf16)(s * u[i]);
}

__global__ void lora_zw_kernel(const float* __restrict__ z,
                               const float* __restrict__ Wd,
                               bf16* __restrict__ zw, size_t n) {
  using namespace cfg;
  size_t i = (size_t)blockIdx.x * blockDim.x + threadIdx.x;
  if (i >= n) return;
  const int t = (int)(i / (E * R));
  const int e = (int)((i % (E * R)) / R);
  zw[i] = (bf16)(z[i] * Wd[(size_t)t * E + e] * LORA_SCALE);
}

__global__ void combine_kernel(const float* __restrict__ h,
                               const float* __restrict__ mlp,
                               const float* __restrict__ swsum,
                               const float* __restrict__ delta,
                               float* __restrict__ out, size_t n) {
  using namespace cfg;
  size_t i = (size_t)blockIdx.x * blockDim.x + threadIdx.x;
  if (i >= n) return;
  const int t = (int)(i / D);
  out[i] = h[i] + mlp[i] * swsum[t] + delta[i];
}

// ------------------------------- launcher ----------------------------------

extern "C" void kernel_launch(void* const* d_in, const int* in_sizes, int n_in,
                              void* d_out, int out_size, void* d_ws, size_t ws_size,
                              hipStream_t stream) {
  using namespace cfg;
  const float* hidden   = (const float*)d_in[0];
  const float* noise    = (const float*)d_in[1];
  const float* ln1_w    = (const float*)d_in[2];
  const float* ln2_w    = (const float*)d_in[3];
  const float* q_w      = (const float*)d_in[4];
  const float* q_b      = (const float*)d_in[5];
  const float* k_w      = (const float*)d_in[6];
  const float* k_b      = (const float*)d_in[7];
  const float* v_w      = (const float*)d_in[8];
  const float* v_b      = (const float*)d_in[9];
  const float* o_w      = (const float*)d_in[10];
  const float* gate_w   = (const float*)d_in[11];
  const float* up_w     = (const float*)d_in[12];
  const float* down_w   = (const float*)d_in[13];
  const float* router_w = (const float*)d_in[14];
  const float* noise_w  = (const float*)d_in[15];
  const float* lora_A   = (const float*)d_in[16];
  const float* lora_B   = (const float*)d_in[17];
  float* out = (float*)d_out;

  char* ws = (char*)d_ws;
  size_t off = 0;
  auto alloc = [&](size_t bytes) {
    size_t o = off;
    off += (bytes + 255) & ~(size_t)255;
    return o;
  };

  // bf16 weight copies
  bf16* qw_b  = (bf16*)(ws + alloc((size_t)D * D * 2));
  bf16* kw_b  = (bf16*)(ws + alloc((size_t)KV * HD * D * 2));
  bf16* vw_b  = (bf16*)(ws + alloc((size_t)KV * HD * D * 2));
  bf16* ow_b  = (bf16*)(ws + alloc((size_t)D * D * 2));
  bf16* gw_b  = (bf16*)(ws + alloc((size_t)FF * D * 2));
  bf16* uw_b  = (bf16*)(ws + alloc((size_t)FF * D * 2));
  bf16* dw_b  = (bf16*)(ws + alloc((size_t)D * FF * 2));
  bf16* la_b  = (bf16*)(ws + alloc((size_t)E * R * D * 2));
  bf16* lbt_b = (bf16*)(ws + alloc((size_t)D * E * R * 2));
  // activations
  bf16*  xb    = (bf16*)(ws + alloc((size_t)T * D * 2));
  float* qf    = (float*)(ws + alloc((size_t)T * D * 4));       // reused for mlp
  float* kf    = (float*)(ws + alloc((size_t)T * KV * HD * 4)); // kf+vf reused for delta
  float* vf    = (float*)(ws + alloc((size_t)T * KV * HD * 4));
  bf16*  qhb   = (bf16*)(ws + alloc((size_t)T * D * 2));
  bf16*  khb   = (bf16*)(ws + alloc((size_t)T * KV * HD * 2));
  bf16*  vhb   = (bf16*)(ws + alloc((size_t)T * KV * HD * 2));
  bf16*  attnb = (bf16*)(ws + alloc((size_t)T * D * 2));        // reused for z/zw
  float* hbuf  = (float*)(ws + alloc((size_t)T * D * 4));
  bf16*  x2b   = (bf16*)(ws + alloc((size_t)T * D * 2));
  float* Wd    = (float*)(ws + alloc((size_t)T * E * 4));
  float* swsum = (float*)(ws + alloc((size_t)T * 4));
  float* gbuf  = (float*)(ws + alloc((size_t)T * FF * 4));
  float* ubuf  = (float*)(ws + alloc((size_t)T * FF * 4));
  bf16*  act   = (bf16*)(ws + alloc((size_t)T * FF * 2));
  // buffer reuse (lifetimes are disjoint)
  float* mlp   = qf;
  float* delta = kf;                       // spans kf+vf (2 * 16MB = T*D*4)
  float* zbuf  = (float*)attnb;            // T*128 f32 inside attnb region
  bf16*  zwb   = (bf16*)(attnb + (size_t)T * (E * R) * 2);

  auto cvt = [&](const float* src, bf16* dst, size_t n) {
    cvt_f32_to_bf16<<<dim3(2048), dim3(256), 0, stream>>>(src, dst, n);
  };
  auto gemm = [&](const bf16* A, const bf16* W, const float* bias,
                  const float* addend, float* C, int M, int N, int K) {
    gemm_bf16_wmma<<<dim3(N / 128, M / 128), dim3(256), 0, stream>>>(
        A, W, bias, addend, C, M, N, K);
  };

  // 0) weight conversions
  cvt(q_w,    qw_b, (size_t)D * D);
  cvt(k_w,    kw_b, (size_t)KV * HD * D);
  cvt(v_w,    vw_b, (size_t)KV * HD * D);
  cvt(o_w,    ow_b, (size_t)D * D);
  cvt(gate_w, gw_b, (size_t)FF * D);
  cvt(up_w,   uw_b, (size_t)FF * D);
  cvt(down_w, dw_b, (size_t)D * FF);
  cvt(lora_A, la_b, (size_t)E * R * D);
  transpose_lora_b<<<dim3(((size_t)E * D * R + 255) / 256), dim3(256), 0, stream>>>(
      lora_B, lbt_b);

  // 1) attention block
  rmsnorm_bf16<<<dim3(T), dim3(256), 0, stream>>>(hidden, ln1_w, xb);
  gemm(xb, qw_b, q_b, nullptr, qf, T, D, D);
  gemm(xb, kw_b, k_b, nullptr, kf, T, KV * HD, D);
  gemm(xb, vw_b, v_b, nullptr, vf, T, KV * HD, D);
  rope_q_kernel<<<dim3((T * H * (HD / 2)) / 256), dim3(256), 0, stream>>>(qf, qhb);
  rope_k_kernel<<<dim3((T * KV * (HD / 2)) / 256), dim3(256), 0, stream>>>(kf, khb);
  cast_v_kernel<<<dim3((T * KV * HD) / 256), dim3(256), 0, stream>>>(vf, vhb);
  flash_attn_wmma<<<dim3(S / 128, H, B), dim3(256), 0, stream>>>(qhb, khb, vhb, attnb);
  gemm(attnb, ow_b, nullptr, hidden, hbuf, T, D, D);  // h = res + attn @ o_w.T

  // 2) MoE block
  rmsnorm_bf16<<<dim3(T), dim3(256), 0, stream>>>(hbuf, ln2_w, x2b);
  router_kernel<<<dim3(T), dim3(256), 0, stream>>>(x2b, router_w, noise_w, noise,
                                                   Wd, swsum);
  gemm(x2b, gw_b, nullptr, nullptr, gbuf, T, FF, D);
  gemm(x2b, uw_b, nullptr, nullptr, ubuf, T, FF, D);
  swiglu_kernel<<<dim3(((size_t)T * FF + 255) / 256), dim3(256), 0, stream>>>(
      gbuf, ubuf, act, (size_t)T * FF);
  gemm(act, dw_b, nullptr, nullptr, mlp, T, D, FF);
  gemm(x2b, la_b, nullptr, nullptr, zbuf, T, E * R, D);
  lora_zw_kernel<<<dim3(((size_t)T * E * R + 255) / 256), dim3(256), 0, stream>>>(
      zbuf, Wd, zwb, (size_t)T * E * R);
  gemm(zwb, lbt_b, nullptr, nullptr, delta, T, D, E * R);
  combine_kernel<<<dim3(((size_t)T * D + 255) / 256), dim3(256), 0, stream>>>(
      hbuf, mlp, swsum, delta, out, (size_t)T * D);

  (void)in_sizes; (void)n_in; (void)out_size; (void)ws_size;
}
